// MultiheadCosineAttention2D_34050500723412
// MI455X (gfx1250) — matrix operand order repaired and checked by reading, not compiled, not run on hardware
//
#include <hip/hip_runtime.h>
#include <hip/hip_bf16.h>

// ---------------------------------------------------------------------------
// Swin-V2 window attention, fused, bf16 WMMA (gfx1250 / CDNA5, wave32)
//   DIM=256, HEADS=8, head_dim=32, N=64 tokens/window, B=2048 windows
//   All-head scoring + single parallel softmax phase (6 barriers total).
// ---------------------------------------------------------------------------

#define DIM     256
#define HEADS   8
#define HD      32
#define NTOK    64
#define NWIN    2048
#define LOG_MAX 4.6051702f   // log(1/0.01)

typedef __attribute__((ext_vector_type(16))) __bf16 bf16x16;
typedef __attribute__((ext_vector_type(8)))  __bf16 bf16x8;
typedef __attribute__((ext_vector_type(4)))  __bf16 bf16x4;
typedef __attribute__((ext_vector_type(8)))  float  f32x8;

typedef float  __attribute__((may_alias)) f32_alias;
typedef __bf16 __attribute__((may_alias)) bf16_alias;

// -------- workspace layout (bytes) --------
#define WS_WQKV   0                      // bf16 [768][256]  = 393216 B
#define WS_WPROJ  393216                 // bf16 [256][256]  = 131072 B
#define WS_BIAS   524288                 // f32  [8][64][64] = 131072 B

// -------- dynamic LDS layout (bytes) --------
#define L_X    0                         // bf16 [64][256] (x, later out1)   32768
#define L_QK   32768                     // bf16 [64][512] (q|k)             65536
#define L_VT   98304                     // bf16 [256][64] (v transposed)    32768
#define L_S    131072                    // f32  [8][64][64] scores          131072
                                         //   (bf16 P written in-place, row stride 128 bf16)
#define L_QN   262144                    // f32  [8][64]                      2048
#define L_KN   264192                    // f32  [8][64]                      2048
#define L_SC   266240                    // f32  [8]                            32
#define SMEM_BYTES 266272

// ---------------------------------------------------------------------------
// WMMA helpers (ISA 7.12.2 VGPR layouts)
// ---------------------------------------------------------------------------
__device__ __forceinline__ f32x8 wmma_bf16(bf16x16 a, bf16x16 b, f32x8 c) {
    return __builtin_amdgcn_wmma_f32_16x16x32_bf16(
        false, a, false, b, (short)0, c, false, false);
}

// A-matrix 16x32 bf16: lane<16 row M=lane, elems 0..7 -> K=0..7, 8..15 -> K=16..23;
// lane>=16 shifts K by +8.
__device__ __forceinline__ bf16x16 load_frag_A(const __bf16* base, int rowStride, int lane) {
    int m = lane & 15, half = lane >> 4;
    const __bf16* p = base + m * rowStride + half * 8;
    union { bf16x16 v; bf16x8 h[2]; } u;
    u.h[0] = *(const bf16x8*)(p);
    u.h[1] = *(const bf16x8*)(p + 16);
    return u.v;
}

// B-matrix 32x16 bf16 stored column-major (storage[n*colStride + k], k contig):
// lane holds column N=lane&15, elems 0..7 -> K=8*(lane>>4)+0..7, 8..15 -> K=16+8*(lane>>4)+0..7.
__device__ __forceinline__ bf16x16 load_frag_B(const __bf16* base, int colStride, int lane) {
    int n = lane & 15, half = lane >> 4;
    const __bf16* p = base + n * colStride + half * 8;
    union { bf16x16 v; bf16x8 h[2]; } u;
    u.h[0] = *(const bf16x8*)(p);
    u.h[1] = *(const bf16x8*)(p + 16);
    return u.v;
}

// ---------------------------------------------------------------------------
// Kernel 1: f32 -> bf16 weight conversion
// ---------------------------------------------------------------------------
__global__ void k_convert_weights(const float* __restrict__ qkv_w,
                                  const float* __restrict__ proj_w,
                                  __bf16* __restrict__ wqkv,
                                  __bf16* __restrict__ wproj) {
    int i = blockIdx.x * 256 + threadIdx.x;
    if (i < 3 * DIM * DIM) wqkv[i]  = (__bf16)qkv_w[i];
    if (i < DIM * DIM)     wproj[i] = (__bf16)proj_w[i];
}

// ---------------------------------------------------------------------------
// Kernel 2: continuous position bias (2 -> 512 -> 8 MLP, gather, 16*sigmoid)
// ---------------------------------------------------------------------------
__global__ void k_cpb_bias(const float* __restrict__ coords,   // [225][2]
                           const float* __restrict__ w1,       // [2][512]
                           const float* __restrict__ b1,       // [512]
                           const float* __restrict__ w2,       // [512][8]
                           const int*   __restrict__ rel_idx,  // [64][64]
                           float* __restrict__ bias_out) {     // [8][64][64]
    __shared__ float table[225 * 8];
    int tid = threadIdx.x;
    for (int r = tid; r < 225; r += 256) {
        float c0 = coords[r * 2 + 0], c1 = coords[r * 2 + 1];
        float acc[8];
        #pragma unroll
        for (int h = 0; h < 8; ++h) acc[h] = 0.f;
        for (int j = 0; j < 512; ++j) {
            float hv = fmaxf(c0 * w1[j] + c1 * w1[512 + j] + b1[j], 0.f);
            #pragma unroll
            for (int h = 0; h < 8; ++h) acc[h] += hv * w2[j * 8 + h];
        }
        #pragma unroll
        for (int h = 0; h < 8; ++h) table[r * 8 + h] = acc[h];
    }
    __syncthreads();
    for (int i = tid; i < HEADS * NTOK * NTOK; i += 256) {
        int h = i >> 12, nm = i & 4095;
        float t = table[rel_idx[nm] * 8 + h];
        bias_out[i] = 16.f / (1.f + __expf(-t));
    }
}

// ---------------------------------------------------------------------------
// Kernel 3: fused window attention. 1 block = 1 window, 256 threads = 8 waves.
// ---------------------------------------------------------------------------
__global__ __launch_bounds__(256)
void k_fused_attn(const float* __restrict__ x,          // [B][64][256]
                  const float* __restrict__ q_bias,     // [256]
                  const float* __restrict__ v_bias,     // [256]
                  const float* __restrict__ logit_scale,// [8]
                  const float* __restrict__ proj_b,     // [256]
                  const __bf16* __restrict__ wqkv,      // [768][256] bf16
                  const __bf16* __restrict__ wproj,     // [256][256] bf16
                  const float* __restrict__ bias,       // [8][64][64]
                  float* __restrict__ out) {            // [B][64][256]
    extern __shared__ char smem[];
    __bf16* x_l  = (__bf16*)(smem + L_X);    // reused as out1 after GEMM1
    __bf16* o1_l = (__bf16*)(smem + L_X);
    __bf16* qk_l = (__bf16*)(smem + L_QK);
    __bf16* vt_l = (__bf16*)(smem + L_VT);
    float*  S_l  = (float*)(smem + L_S);     // [512][64] f32 (g = h*64+n)
    float*  qn_l = (float*)(smem + L_QN);
    float*  kn_l = (float*)(smem + L_KN);
    float*  sc_l = (float*)(smem + L_SC);

    const int b    = blockIdx.x;
    const int tid  = threadIdx.x;
    const int lane = tid & 31;
    const int wave = tid >> 5;

    // ---- stage x (f32 -> bf16) into LDS ----
    const float* xb = x + (size_t)b * NTOK * DIM;
    for (int i = tid * 4; i < NTOK * DIM; i += 1024) {
        float4 f = *(const float4*)(xb + i);
        bf16x4 o;
        o[0] = (__bf16)f.x; o[1] = (__bf16)f.y;
        o[2] = (__bf16)f.z; o[3] = (__bf16)f.w;
        *(bf16x4*)(x_l + i) = o;
    }
    if (tid < HEADS) sc_l[tid] = __expf(fminf(logit_scale[tid], LOG_MAX));
    __syncthreads();

    // ---- GEMM1: qkv[64][768] = x[64][256] @ wqkv^T, tiles 4m x 48n ----
    for (int tm = 0; tm < 4; ++tm) {
        bf16x16 Areg[8];
        #pragma unroll
        for (int kc = 0; kc < 8; ++kc)
            Areg[kc] = load_frag_A(x_l + tm * 16 * DIM + kc * 32, DIM, lane);
        for (int j = 0; j < 6; ++j) {
            int tn = wave + 8 * j;                   // 0..47
            f32x8 c = {};
            const __bf16* wb = wqkv + (size_t)(tn * 16) * DIM;
            #pragma unroll
            for (int kc = 0; kc < 8; ++kc) {
                bf16x16 B = load_frag_B(wb + kc * 32, DIM, lane);
                c = wmma_bf16(Areg[kc], B, c);
            }
            int coln  = tn * 16 + (lane & 15);       // qkv channel 0..767
            float bv  = (coln < 256) ? q_bias[coln]
                      : (coln >= 512 ? v_bias[coln - 512] : 0.f);
            int rbase = tm * 16 + 8 * (lane >> 4);
            #pragma unroll
            for (int r = 0; r < 8; ++r) {
                float v = c[r] + bv;
                int row = rbase + r;
                if (coln < 512) qk_l[row * 512 + coln] = (__bf16)v;        // q|k
                else            vt_l[(coln - 512) * 64 + row] = (__bf16)v; // v^T
            }
        }
    }
    __syncthreads();

    // ---- row norms: 1/|q_{h,n}|, 1/|k_{h,n}| (vectorized 16B LDS loads) ----
    for (int idx = tid; idx < HEADS * NTOK; idx += 256) {
        int h = idx >> 6, n = idx & 63;
        const bf16x8* qr = (const bf16x8*)(qk_l + n * 512 + h * HD);
        const bf16x8* kr = (const bf16x8*)(qk_l + n * 512 + 256 + h * HD);
        float sq = 0.f, sk = 0.f;
        #pragma unroll
        for (int j = 0; j < 4; ++j) {
            bf16x8 qv = qr[j], kv = kr[j];
            #pragma unroll
            for (int e = 0; e < 8; ++e) {
                float qf = (float)qv[e]; sq += qf * qf;
                float kf = (float)kv[e]; sk += kf * kf;
            }
        }
        qn_l[idx] = rsqrtf(sq);
        kn_l[idx] = rsqrtf(sk);
    }
    __syncthreads();

    // ---- all-head scores: S[h][64][64] = q_h @ k_h^T, wave w owns head w ----
    {
        const int h = wave;
        bf16x16 Aq[4], Bk[4];
        #pragma unroll
        for (int tm = 0; tm < 4; ++tm)
            Aq[tm] = load_frag_A(qk_l + tm * 16 * 512 + h * HD, 512, lane);
        #pragma unroll
        for (int tn = 0; tn < 4; ++tn)
            Bk[tn] = load_frag_B(qk_l + tn * 16 * 512 + 256 + h * HD, 512, lane);
        #pragma unroll
        for (int tm = 0; tm < 4; ++tm) {
            #pragma unroll
            for (int tn = 0; tn < 4; ++tn) {
                f32x8 c = {};
                c = wmma_bf16(Aq[tm], Bk[tn], c);
                int coln  = tn * 16 + (lane & 15);
                int rbase = h * 64 + tm * 16 + 8 * (lane >> 4);
                #pragma unroll
                for (int r = 0; r < 8; ++r) S_l[(rbase + r) * 64 + coln] = c[r];
            }
        }
    }
    __syncthreads();

    // ---- softmax: 512 rows, 2 per thread; bf16 P written in-place over S ----
    for (int g = tid; g < HEADS * NTOK; g += 256) {
        int h = g >> 6;
        float rq = qn_l[g] * sc_l[h];
        f32_alias*  Sr = (f32_alias*)(smem + L_S) + g * 64;
        bf16_alias* Pr = (bf16_alias*)Sr;             // row stride 128 bf16
        const float* bs = bias + g * 64;              // g*64 == h*4096 + n*64
        float mx = -1e30f;
        for (int m = 0; m < 64; ++m) {
            float v = Sr[m] * (rq * kn_l[h * 64 + m]) + bs[m];
            Sr[m] = v;
            mx = fmaxf(mx, v);
        }
        float s = 0.f;
        for (int m = 0; m < 64; ++m) {
            float e = __expf(Sr[m] - mx);
            Sr[m] = e;
            s += e;
        }
        float inv = 1.f / s;
        for (int m = 0; m < 64; ++m)                  // ascending m: write@2m
            Pr[m] = (__bf16)(Sr[m] * inv);            // clobbers consumed m/2
    }
    __syncthreads();

    // ---- out_h = P_h @ v_h, wave w owns head w (4m x 2n tiles, K=64) ----
    {
        const int h = wave;
        const __bf16* Pbase = (const __bf16*)(smem + L_S) + h * 64 * 128;
        bf16x16 Ap[4][2], Bv[2][2];
        #pragma unroll
        for (int tm = 0; tm < 4; ++tm)
            #pragma unroll
            for (int kc = 0; kc < 2; ++kc)
                Ap[tm][kc] = load_frag_A(Pbase + tm * 16 * 128 + kc * 32, 128, lane);
        #pragma unroll
        for (int tn = 0; tn < 2; ++tn)
            #pragma unroll
            for (int kc = 0; kc < 2; ++kc)
                Bv[tn][kc] = load_frag_B(vt_l + (h * HD + tn * 16) * 64 + kc * 32, 64, lane);
        #pragma unroll
        for (int tm = 0; tm < 4; ++tm) {
            #pragma unroll
            for (int tn = 0; tn < 2; ++tn) {
                f32x8 c = {};
                c = wmma_bf16(Ap[tm][0], Bv[tn][0], c);
                c = wmma_bf16(Ap[tm][1], Bv[tn][1], c);
                int coln  = h * HD + tn * 16 + (lane & 15);
                int rbase = tm * 16 + 8 * (lane >> 4);
                #pragma unroll
                for (int r = 0; r < 8; ++r)
                    o1_l[(rbase + r) * DIM + coln] = (__bf16)c[r];
            }
        }
    }
    __syncthreads();

    // ---- GEMM2: out[64][256] = out1 @ wproj^T + proj_b, tiles 4m x 16n ----
    float* ob = out + (size_t)b * NTOK * DIM;
    for (int tm = 0; tm < 4; ++tm) {
        bf16x16 Areg[8];
        #pragma unroll
        for (int kc = 0; kc < 8; ++kc)
            Areg[kc] = load_frag_A(o1_l + tm * 16 * DIM + kc * 32, DIM, lane);
        #pragma unroll
        for (int j = 0; j < 2; ++j) {
            int tn = wave + 8 * j;                   // 0..15
            f32x8 c = {};
            const __bf16* wb = wproj + (size_t)(tn * 16) * DIM;
            #pragma unroll
            for (int kc = 0; kc < 8; ++kc) {
                bf16x16 B = load_frag_B(wb + kc * 32, DIM, lane);
                c = wmma_bf16(Areg[kc], B, c);
            }
            int coln  = tn * 16 + (lane & 15);
            float pb  = proj_b[coln];
            int rbase = tm * 16 + 8 * (lane >> 4);
            #pragma unroll
            for (int r = 0; r < 8; ++r)
                ob[(rbase + r) * DIM + coln] = c[r] + pb;
        }
    }
}

// ---------------------------------------------------------------------------
extern "C" void kernel_launch(void* const* d_in, const int* in_sizes, int n_in,
                              void* d_out, int out_size, void* d_ws, size_t ws_size,
                              hipStream_t stream) {
    const float* x           = (const float*)d_in[0];
    const float* qkv_w       = (const float*)d_in[1];
    const float* q_bias      = (const float*)d_in[2];
    const float* v_bias      = (const float*)d_in[3];
    const float* logit_scale = (const float*)d_in[4];
    const float* cpb_w1      = (const float*)d_in[5];
    const float* cpb_b1      = (const float*)d_in[6];
    const float* cpb_w2      = (const float*)d_in[7];
    const float* proj_w      = (const float*)d_in[8];
    const float* proj_b      = (const float*)d_in[9];
    const float* coords      = (const float*)d_in[10];
    const int*   rel_idx     = (const int*)d_in[11];

    char*   ws    = (char*)d_ws;
    __bf16* wqkv  = (__bf16*)(ws + WS_WQKV);
    __bf16* wproj = (__bf16*)(ws + WS_WPROJ);
    float*  biasb = (float*)(ws + WS_BIAS);

    k_convert_weights<<<(3 * DIM * DIM + 255) / 256, 256, 0, stream>>>(
        qkv_w, proj_w, wqkv, wproj);
    k_cpb_bias<<<1, 256, 0, stream>>>(coords, cpb_w1, cpb_b1, cpb_w2, rel_idx, biasb);
    k_fused_attn<<<NWIN, 256, SMEM_BYTES, stream>>>(
        x, q_bias, v_bias, logit_scale, proj_b, wqkv, wproj, biasb, (float*)d_out);
}
